// MultiGenPool_13907104104776
// MI455X (gfx1250) — matrix operand order repaired and checked by reading, not compiled, not run on hardware
//
#include <hip/hip_runtime.h>
#include <hip/hip_bf16.h>
#include <math.h>

typedef __attribute__((ext_vector_type(16))) __bf16 v16bf;
typedef __attribute__((ext_vector_type(8)))  float  v8f;

namespace {
constexpr int Bn = 8;
constexpr int Sn = 4096;
constexpr int Dn = 1024;
constexpr int Hn = 8;
constexpr int DHn = 128;
constexpr int DOn = 128;
constexpr float kNegInf = 32752.0f;

constexpr int SAF_STR = 36;  // fp32 feature tile row stride (36 floats = 144B,
                             // 16B aligned rows + bank-spread: 36 dw mod 64)
constexpr int SBT_STR = 36;  // bf16 transposed w1 tile row stride (halfs)

// XOR swizzle on 16-element column groups: contiguous 8/16-half runs that
// start 8/16-aligned stay contiguous, while rows land on distinct banks.
__device__ __forceinline__ int swz(int row, int col) {
  return row * 128 + (col ^ ((row & 7) << 4));
}

// CDNA5 async global->LDS copy, 16B per lane, tracked by ASYNCcnt.
__device__ __forceinline__ void async_copy_b128(unsigned lds_byte_addr,
                                                const void* gaddr) {
  asm volatile("global_load_async_to_lds_b128 %0, %1, off"
               :: "v"(lds_byte_addr), "v"(gaddr) : "memory");
}
__device__ __forceinline__ void wait_async0() {
  asm volatile("s_wait_asynccnt 0x0" ::: "memory");
}
} // namespace

// ---------------------------------------------------------------------------
// Kernel 1: per (b, h, 128-row seq tile) compute
//   scores = GELU(feat @ w1h + b1h) @ w2h + b2h, with mask -> -32752
// bf16 WMMA (fp32 accumulate). 8 waves, each owns a 16x128 strip.
// Feature tiles arrive via GLOBAL_LOAD_ASYNC_TO_LDS_B128 (ASYNCcnt).
// ---------------------------------------------------------------------------
__global__ __launch_bounds__(256)
void k_scores(const float* __restrict__ feat,
              const unsigned char* __restrict__ mask,
              const float* __restrict__ w1, const float* __restrict__ b1,
              const float* __restrict__ w2, const float* __restrict__ b2,
              float* __restrict__ scores)
{
  __shared__ __align__(128) char smem[65536];
  float*  sAf = (float*)smem;               // phase A: [128][SAF_STR] fp32
  __bf16* sBt = (__bf16*)(smem + 18432);    // phase A: [128 cols][SBT_STR] bf16
  __bf16* sWt = (__bf16*)(smem + 32768);    // persistent: swizzled [128][128]
  __bf16* sH  = (__bf16*)smem;              // phase B: swizzled [128][128]

  const int tid  = threadIdx.x;
  const int lane = tid & 31;
  const int wv   = tid >> 5;   // wave id 0..7 -> 16-row strip
  const int lr   = lane & 15;
  const int lhi  = lane >> 4;

  const int stiles = Sn / 128;
  const int st = blockIdx.x % stiles;
  const int h  = (blockIdx.x / stiles) % Hn;
  const int b  = blockIdx.x / (stiles * Hn);
  const int s0 = st * 128;

  const float* featb = feat + (size_t)b * Sn * Dn;
  const float* w1h   = w1 + (size_t)h * Dn * DHn;
  const float* w2h   = w2 + (size_t)h * DHn * DOn;

  // Stage w2 head transposed+swizzled into the upper 32KB (not aliased by
  // phase A, so it can be loaded once up front). float4 global reads.
  for (int i = tid; i < (DHn * DOn) / 4; i += 256) {
    const int k  = i >> 5;          // 32 float4 per k-row
    const int c4 = (i & 31) * 4;
    const float4 w = *(const float4*)&w2h[(size_t)k * DOn + c4];
    sWt[swz(c4 + 0, k)] = (__bf16)w.x;
    sWt[swz(c4 + 1, k)] = (__bf16)w.y;
    sWt[swz(c4 + 2, k)] = (__bf16)w.z;
    sWt[swz(c4 + 3, k)] = (__bf16)w.w;
  }

  v8f acc[8] = {};

  for (int kb = 0; kb < Dn; kb += 32) {
    // Async-copy the 128x32 fp32 feature tile straight into LDS
    // (no VGPR round-trip; 16B per lane per op).
    for (int i = tid; i < 128 * 8; i += 256) {
      const int r  = i >> 3;
      const int c4 = (i & 7) * 4;
      const unsigned lds = (unsigned)(uintptr_t)&sAf[r * SAF_STR + c4];
      async_copy_b128(lds, &featb[(size_t)(s0 + r) * Dn + kb + c4]);
    }
    // w1 32x128 tile -> transposed bf16 LDS (direct path; needs transpose)
    for (int i = tid; i < 32 * 32; i += 256) {
      const int r  = i >> 5;
      const int c4 = (i & 31) * 4;
      const float4 w = *(const float4*)&w1h[(size_t)(kb + r) * DHn + c4];
      sBt[(c4 + 0) * SBT_STR + r] = (__bf16)w.x;
      sBt[(c4 + 1) * SBT_STR + r] = (__bf16)w.y;
      sBt[(c4 + 2) * SBT_STR + r] = (__bf16)w.z;
      sBt[(c4 + 3) * SBT_STR + r] = (__bf16)w.w;
    }
    if (kb + 32 < Dn) {  // hint next K tile into cache (global_prefetch_b8)
      __builtin_prefetch(&featb[(size_t)(s0 + (tid >> 1)) * Dn + kb + 32], 0, 1);
      __builtin_prefetch(&w1h[(size_t)(kb + 32 + (tid >> 3)) * DHn + ((tid & 7) << 4)], 0, 1);
    }
    wait_async0();       // per-wave ASYNCcnt drain before publishing
    __syncthreads();

    // A fragment (ISA 16-bit A 16x32 layout: lane=M, K 8-granular by lane
    // half). Read fp32 rows, convert to bf16 at fragment build.
    v16bf a;
    {
      const float* pr = &sAf[(wv * 16 + lr) * SAF_STR];
      const int kbh = lhi * 8;
      const float4* q0 = (const float4*)&pr[kbh];
      const float4* q1 = (const float4*)&pr[16 + kbh];
      const float4 x0 = q0[0], x1 = q0[1];
      const float4 y0 = q1[0], y1 = q1[1];
      a[0]  = (__bf16)x0.x; a[1]  = (__bf16)x0.y;
      a[2]  = (__bf16)x0.z; a[3]  = (__bf16)x0.w;
      a[4]  = (__bf16)x1.x; a[5]  = (__bf16)x1.y;
      a[6]  = (__bf16)x1.z; a[7]  = (__bf16)x1.w;
      a[8]  = (__bf16)y0.x; a[9]  = (__bf16)y0.y;
      a[10] = (__bf16)y0.z; a[11] = (__bf16)y0.w;
      a[12] = (__bf16)y1.x; a[13] = (__bf16)y1.y;
      a[14] = (__bf16)y1.z; a[15] = (__bf16)y1.w;
    }
    #pragma unroll
    for (int n = 0; n < 8; ++n) {
      // B fragment (lane=N, K 8-granular by lane half, same as A)
      v16bf bq;
      const __bf16* p = &sBt[(n * 16 + lr) * SBT_STR + lhi * 8];
      #pragma unroll
      for (int j = 0; j < 8; ++j) { bq[j] = p[j]; bq[8 + j] = p[16 + j]; }
      acc[n] = __builtin_amdgcn_wmma_f32_16x16x32_bf16(
          false, a, false, bq, (short)0, acc[n], false, false);
    }
    __syncthreads();
  }

  // bias + exact GELU on fp32 accumulators, deposit h strip (swizzled bf16).
  {
    const float* b1h = b1 + h * DHn;
    #pragma unroll
    for (int n = 0; n < 8; ++n) {
      const float bias = b1h[n * 16 + lr];
      #pragma unroll
      for (int v = 0; v < 8; ++v) {
        float x = acc[n][v] + bias;                       // C layout: N=lr
        x = 0.5f * x * (1.0f + erff(x * 0.70710678118654752f));
        const int row = wv * 16 + lhi * 8 + v;            // C layout: M
        sH[swz(row, n * 16 + lr)] = (__bf16)x;
      }
    }
  }
  __syncthreads();

  // Second GEMM: scores strip (16x128) = h strip (16x128) @ w2 (128x128)
  v8f acc2[8] = {};
  #pragma unroll
  for (int kt = 0; kt < 4; ++kt) {
    v16bf a;
    {
      const int r = wv * 16 + lr;
      const __bf16* p0 = &sH[swz(r, kt * 32 + lhi * 8)];
      const __bf16* p1 = &sH[swz(r, kt * 32 + 16 + lhi * 8)];
      #pragma unroll
      for (int j = 0; j < 8; ++j) { a[j] = p0[j]; a[8 + j] = p1[j]; }
    }
    #pragma unroll
    for (int n = 0; n < 8; ++n) {
      v16bf bq;
      const __bf16* p = &sWt[swz(n * 16 + lr, kt * 32 + lhi * 8)];
      const __bf16* q = &sWt[swz(n * 16 + lr, kt * 32 + 16 + lhi * 8)];
      #pragma unroll
      for (int j = 0; j < 8; ++j) { bq[j] = p[j]; bq[8 + j] = q[j]; }
      acc2[n] = __builtin_amdgcn_wmma_f32_16x16x32_bf16(
          false, a, false, bq, (short)0, acc2[n], false, false);
    }
  }

  // b2, mask (-32752), write scores (b,h,s,o)
  {
    const float* b2h = b2 + h * DOn;
    const unsigned char* mrow = mask + (size_t)b * Sn;
    float* scb = scores + (size_t)(b * Hn + h) * Sn * DOn;
    #pragma unroll
    for (int n = 0; n < 8; ++n) {
      const float bias = b2h[n * 16 + lr];
      #pragma unroll
      for (int v = 0; v < 8; ++v) {
        const int s = s0 + wv * 16 + lhi * 8 + v;
        float x = acc2[n][v] + bias;
        if (mrow[s]) x = -kNegInf;
        scb[(size_t)s * DOn + n * 16 + lr] = x;
      }
    }
  }
}

// ---------------------------------------------------------------------------
// Kernel 2: per-(b,h,o) column softmax stats over S (max + sum of exp)
// ---------------------------------------------------------------------------
__global__ __launch_bounds__(256)
void k_stats(const float* __restrict__ scores,
             float* __restrict__ maxv, float* __restrict__ sumv)
{
  __shared__ float red[256];
  const int c  = blockIdx.x;          // c = b*1024 + h*128 + o
  const int b  = c >> 10;
  const int ho = c & 1023;
  const int h  = ho >> 7, o = ho & 127;
  const float* col = scores + (size_t)(b * Hn + h) * Sn * DOn + o;
  const int t = threadIdx.x;

  float m = -3.4e38f;
  for (int s = t; s < Sn; s += 256) m = fmaxf(m, col[(size_t)s * DOn]);
  red[t] = m; __syncthreads();
  for (int off = 128; off > 0; off >>= 1) {
    if (t < off) red[t] = fmaxf(red[t], red[t + off]);
    __syncthreads();
  }
  m = red[0]; __syncthreads();

  float sm = 0.0f;
  for (int s = t; s < Sn; s += 256) sm += __expf(col[(size_t)s * DOn] - m);
  red[t] = sm; __syncthreads();
  for (int off = 128; off > 0; off >>= 1) {
    if (t < off) red[t] += red[t + off];
    __syncthreads();
  }
  if (t == 0) { maxv[c] = m; sumv[c] = red[0]; }
}

// ---------------------------------------------------------------------------
// Kernel 3: pooled(b,d) = sum_s feat(b,s,d) * softmax_weight(b,h,s,o)
// ---------------------------------------------------------------------------
__global__ __launch_bounds__(256)
void k_pool(const float* __restrict__ feat, const float* __restrict__ scores,
            const float* __restrict__ maxv, const float* __restrict__ sumv,
            float* __restrict__ out)
{
  const int b = blockIdx.x >> 2;                       // Dn/256 = 4 chunks
  const int d = ((blockIdx.x & 3) << 8) + threadIdx.x; // 0..1023
  const int h = d >> 7, o = d & 127;
  const int c = b * Dn + d;
  const float m   = maxv[c];
  const float inv = 1.0f / sumv[c];
  const float* f  = feat + (size_t)b * Sn * Dn + d;
  const float* sc = scores + (size_t)(b * Hn + h) * Sn * DOn + o;
  float acc = 0.0f;
  for (int s = 0; s < Sn; ++s) {
    const float w = __expf(sc[(size_t)s * DOn] - m) * inv;
    acc += f[(size_t)s * Dn] * w;
  }
  out[b * Dn + d] = acc;
}

// ---------------------------------------------------------------------------
extern "C" void kernel_launch(void* const* d_in, const int* in_sizes, int n_in,
                              void* d_out, int out_size, void* d_ws, size_t ws_size,
                              hipStream_t stream)
{
  (void)in_sizes; (void)n_in; (void)out_size; (void)ws_size;
  const float*         feat = (const float*)d_in[0];
  const unsigned char* mask = (const unsigned char*)d_in[1];
  // d_in[2] = lengths: unused by the reference
  const float* w1 = (const float*)d_in[3];
  const float* b1 = (const float*)d_in[4];
  const float* w2 = (const float*)d_in[5];
  const float* b2 = (const float*)d_in[6];
  float* out = (float*)d_out;

  // Workspace layout: scores (B*H*S*DO fp32 = 128MB), then max/sum stats.
  float* scores = (float*)d_ws;
  const size_t nsc = (size_t)Bn * Hn * Sn * DOn;
  float* maxv = scores + nsc;
  float* sumv = maxv + (size_t)Bn * Hn * DOn;

  k_scores<<<dim3(Bn * Hn * (Sn / 128)), 256, 0, stream>>>(
      feat, mask, w1, b1, w2, b2, scores);
  k_stats<<<dim3(Bn * Hn * DOn), 256, 0, stream>>>(scores, maxv, sumv);
  k_pool<<<dim3(Bn * (Dn / 256)), 256, 0, stream>>>(
      feat, scores, maxv, sumv, out);
}
